// SparseResNet_5798205849717
// MI455X (gfx1250) — matrix operand order repaired and checked by reading, not compiled, not run on hardware
//
#include <hip/hip_runtime.h>

// ---------------------------------------------------------------------------
// Types
// ---------------------------------------------------------------------------
typedef __bf16 bf16_t;
typedef __attribute__((ext_vector_type(16))) __bf16        v16bf;
typedef __attribute__((ext_vector_type(8)))  float         v8f;
typedef __attribute__((ext_vector_type(4)))  unsigned int  v4u;

union Frag256 {
    v4u   u[2];
    v16bf v;
};

// gfx1250 async global->LDS copy path (guarded; falls back to reg-staged copy)
#if defined(__gfx1250__) && __has_builtin(__builtin_amdgcn_global_load_async_to_lds_b128) && \
    __has_builtin(__builtin_amdgcn_s_wait_asynccnt)
#define USE_ASYNC_LDS 1
// builtin expects: (int __vector(4) __device__* src, int __vector(4) __shared__* dst,
//                   imm offset, imm cpol)
typedef int v4i_g __attribute__((vector_size(16)));
typedef __attribute__((address_space(1))) v4i_g as1_v4i;
typedef __attribute__((address_space(3))) v4i_g as3_v4i;
#else
#define USE_ASYNC_LDS 0
#endif

// ---------------------------------------------------------------------------
// Fused gather + 27-offset GEMM using v_wmma_f32_16x16x32_bf16, with
// double-buffered LDS software pipelining across the 27 kernel offsets.
//   act : [Nin,  CIN]     bf16 (source features)
//   nb  : [27,   M]       int32 rulebook (row index into act, or -1)
//   wt  : [27, COUT, CIN] bf16 (weights pre-transposed to N-major)
//   out : [M, COUT]       f32
// Workgroup = 256 threads (8 wave32), computes a BM x COUT output tile.
// ---------------------------------------------------------------------------
template<int CIN, int COUT, int BM>
__global__ __launch_bounds__(256)
void spconv_wmma(const bf16_t* __restrict__ act,
                 const int*    __restrict__ nb,
                 const bf16_t* __restrict__ wt,
                 float*        __restrict__ out,
                 int M)
{
    constexpr int PAD    = 8;                // 16B pad/row -> conflict-free ds_b128
    constexpr int LDA    = CIN + PAD;        // LDS row pitch (bf16 elements)
    constexpr int NT     = COUT / 16;        // N tiles
    constexpr int RT     = BM / 16;          // M tiles
    constexpr int NWAVES = 8;
    constexpr int TPW    = (NT * RT) / NWAVES;
    static_assert((NT * RT) % NWAVES == 0, "tiles must divide waves");
    static_assert(RT % TPW == 0, "wave's tiles must share one N strip");

    constexpr int BUFELEMS = BM * LDA;
    __shared__ __align__(16) bf16_t As[2 * BUFELEMS];

    const int m0    = blockIdx.x * BM;
    const int tid   = threadIdx.x;
    const int wave  = tid >> 5;
    const int lane  = tid & 31;
    const int lhalf = lane >> 4;             // 0 or 1
    const int lmod  = lane & 15;

    v8f acc[TPW];
#pragma unroll
    for (int t = 0; t < TPW; ++t) acc[t] = v8f{};

    const int nt = (wave * TPW) / RT;        // N strip owned by this wave

    constexpr int CHUNKS = CIN / 8;          // 16B chunks per row
    constexpr int ITERS  = (BM * CHUNKS) / 256;

#if USE_ASYNC_LDS
    // Issue per-lane async copies directly into LDS; zero-fill masked rows.
    auto gather_async = [&](int k, bf16_t* dst) {
#pragma unroll
        for (int j = 0; j < ITERS; ++j) {
            const int linear = tid + j * 256;
            const int row    = linear / CHUNKS;
            const int ch     = linear % CHUNKS;
            const int src    = nb[(size_t)k * M + m0 + row];
            bf16_t* lp = dst + row * LDA + ch * 8;
            if (src >= 0) {
                __builtin_amdgcn_global_load_async_to_lds_b128(
                    (as1_v4i*)(act + (size_t)src * CIN + ch * 8),
                    (as3_v4i*)lp, 0, 0);
            } else {
                *(v4u*)lp = (v4u)0;
            }
        }
    };
#else
    v4u vals[ITERS];
    auto gather_issue = [&](int k) {
#pragma unroll
        for (int j = 0; j < ITERS; ++j) {
            const int linear = tid + j * 256;
            const int row    = linear / CHUNKS;
            const int ch     = linear % CHUNKS;
            const int src    = nb[(size_t)k * M + m0 + row];
            v4u v = (v4u)0;
            if (src >= 0)
                v = *(const v4u*)(act + (size_t)src * CIN + ch * 8);
            vals[j] = v;
        }
    };
    auto gather_commit = [&](bf16_t* dst) {
#pragma unroll
        for (int j = 0; j < ITERS; ++j) {
            const int linear = tid + j * 256;
            const int row    = linear / CHUNKS;
            const int ch     = linear % CHUNKS;
            *(v4u*)(dst + row * LDA + ch * 8) = vals[j];
        }
    };
#endif

    // ---- prologue: stage offset 0 into buffer 0 ----
#if USE_ASYNC_LDS
    gather_async(0, As);
    __builtin_amdgcn_s_wait_asynccnt(0);
#else
    gather_issue(0);
    gather_commit(As);
#endif
    __syncthreads();

    for (int k = 0; k < 27; ++k) {
        bf16_t* cur = As + (k & 1) * BUFELEMS;
        bf16_t* nxt = As + ((k + 1) & 1) * BUFELEMS;

        // stage next offset while this one computes
        if (k + 1 < 27) {
#if USE_ASYNC_LDS
            gather_async(k + 1, nxt);
#else
            gather_issue(k + 1);
#endif
            // hint: pull next offset's weight panel toward the caches
            __builtin_prefetch(
                wt + (((size_t)(k + 1) * COUT + nt * 16 + lmod) * CIN), 0, 1);
        }

        // ---- K loop: 32-deep bf16 WMMA steps on current buffer ----
#pragma unroll
        for (int kc = 0; kc < CIN / 32; ++kc) {
            // B fragment: 32x16 bf16, lane<16 -> N=lmod K[0..15], lane>=16 -> K[16..31]
            const bf16_t* bp = wt + (((size_t)k * COUT + nt * 16 + lmod) * CIN)
                                  + kc * 32 + lhalf * 16;
            Frag256 bf;
            bf.u[0] = *(const v4u*)(bp);
            bf.u[1] = *(const v4u*)(bp + 8);

#pragma unroll
            for (int t = 0; t < TPW; ++t) {
                const int rt = (wave * TPW + t) % RT;
                // A fragment: 16x32 bf16, lane<16 -> K[0..15], lane>=16 -> K[16..31]
                const bf16_t* ap = &cur[(rt * 16 + lmod) * LDA + kc * 32 + lhalf * 16];
                Frag256 af;
                af.u[0] = *(const v4u*)(ap);
                af.u[1] = *(const v4u*)(ap + 8);
                acc[t] = __builtin_amdgcn_wmma_f32_16x16x32_bf16(
                    false, af.v, false, bf.v, (short)0, acc[t], false, false);
            }
        }

#if !USE_ASYNC_LDS
        if (k + 1 < 27) gather_commit(nxt);
#else
        __builtin_amdgcn_s_wait_asynccnt(0);
#endif
        __syncthreads();
    }

    // ---- store f32 accumulators (C/D layout: VGPR j -> M=j+8*lhalf, N=lmod) ----
#pragma unroll
    for (int t = 0; t < TPW; ++t) {
        const int id = wave * TPW + t;
        const int rt = id % RT;
        const int n  = (id / RT) * 16 + lmod;
#pragma unroll
        for (int j = 0; j < 8; ++j) {
            const int m = m0 + rt * 16 + j + lhalf * 8;
            out[(size_t)m * COUT + n] = acc[t][j];
        }
    }
}

// ---------------------------------------------------------------------------
// BatchNorm statistics: one block per channel, deterministic tree reduction.
// Writes mean[c] and rsqrt(var[c]+eps) into stats[0..C-1], stats[C..2C-1].
// ---------------------------------------------------------------------------
__global__ __launch_bounds__(256)
void bn_stats(const float* __restrict__ f, int M, int C,
              float* __restrict__ stats)
{
    const int c = blockIdx.x;
    float s = 0.f, sq = 0.f;
    for (int r = threadIdx.x; r < M; r += 256) {
        const float v = f[(size_t)r * C + c];
        s  += v;
        sq += v * v;
    }
    __shared__ float sh0[256];
    __shared__ float sh1[256];
    sh0[threadIdx.x] = s;
    sh1[threadIdx.x] = sq;
    __syncthreads();
    for (int st = 128; st > 0; st >>= 1) {
        if (threadIdx.x < st) {
            sh0[threadIdx.x] += sh0[threadIdx.x + st];
            sh1[threadIdx.x] += sh1[threadIdx.x + st];
        }
        __syncthreads();
    }
    if (threadIdx.x == 0) {
        const float inv = 1.0f / (float)M;
        const float mu  = sh0[0] * inv;
        const float var = sh1[0] * inv - mu * mu;
        stats[c]     = mu;
        stats[C + c] = rsqrtf(var + 1e-5f);
    }
}

// ---------------------------------------------------------------------------
// BN apply + optional residual + ReLU; writes f32 and/or bf16 copies.
// ---------------------------------------------------------------------------
__global__ __launch_bounds__(256)
void bn_apply(const float* __restrict__ f,
              const float* __restrict__ stats,
              const float* __restrict__ g,
              const float* __restrict__ b,
              const float* __restrict__ resid,    // may be null
              float*       __restrict__ out_f32,  // may be null
              bf16_t*      __restrict__ out_bf16, // may be null
              int M, int C)
{
    const int i = blockIdx.x * 256 + threadIdx.x;
    if (i >= M * C) return;
    const int c = i % C;
    float v = (f[i] - stats[c]) * stats[C + c] * g[c] + b[c];
    if (resid) v += resid[i];
    v = fmaxf(v, 0.f);
    if (out_f32)  out_f32[i]  = v;
    if (out_bf16) out_bf16[i] = (bf16_t)v;
}

// ---------------------------------------------------------------------------
// Weight convert + transpose:  w[27][Cin][Cout] f32 -> wt[27][Cout][Cin] bf16
// ---------------------------------------------------------------------------
__global__ __launch_bounds__(256)
void wconv(const float* __restrict__ w, bf16_t* __restrict__ wt,
           int Cin, int Cout)
{
    const int o = blockIdx.x * 256 + threadIdx.x;
    const int total = 27 * Cin * Cout;
    if (o >= total) return;
    const int c = o % Cin;
    const int n = (o / Cin) % Cout;
    const int k = o / (Cin * Cout);
    wt[o] = (bf16_t)w[((size_t)k * Cin + c) * Cout + n];
}

// ---------------------------------------------------------------------------
// f32 -> bf16 copy
// ---------------------------------------------------------------------------
__global__ __launch_bounds__(256)
void f2bf(const float* __restrict__ in, bf16_t* __restrict__ out, int n)
{
    const int i = blockIdx.x * 256 + threadIdx.x;
    if (i < n) out[i] = (bf16_t)in[i];
}

// ---------------------------------------------------------------------------
// Launcher
// ---------------------------------------------------------------------------
extern "C" void kernel_launch(void* const* d_in, const int* in_sizes, int n_in,
                              void* d_out, int out_size, void* d_ws, size_t ws_size,
                              hipStream_t stream)
{
    (void)n_in; (void)out_size; (void)ws_size;

    const float* x    = (const float*)d_in[0];
    const float* w_s1 = (const float*)d_in[1];
    const float* g_s1 = (const float*)d_in[2];
    const float* b_s1 = (const float*)d_in[3];
    const float* w11  = (const float*)d_in[4];
    const float* g11  = (const float*)d_in[5];
    const float* b11  = (const float*)d_in[6];
    const float* w12  = (const float*)d_in[7];
    const float* g12  = (const float*)d_in[8];
    const float* b12  = (const float*)d_in[9];
    const float* w21  = (const float*)d_in[10];
    const float* g21  = (const float*)d_in[11];
    const float* b21  = (const float*)d_in[12];
    const float* w22  = (const float*)d_in[13];
    const float* g22  = (const float*)d_in[14];
    const float* b22  = (const float*)d_in[15];
    const float* w_s2 = (const float*)d_in[16];
    const float* g_s2 = (const float*)d_in[17];
    const float* b_s2 = (const float*)d_in[18];
    const int*   nb1  = (const int*)d_in[19];
    const int*   nbs  = (const int*)d_in[20];
    const int*   nb2  = (const int*)d_in[21];
    float*       outp = (float*)d_out;

    const int N0 = in_sizes[0] / 256;   // 9216
    const int M1 = in_sizes[19] / 27;   // 27648
    const int M2 = in_sizes[21] / 27;   // 46080

    // --- workspace carving (256B aligned) ---
    size_t off = 0;
    auto carve = [&](size_t bytes) -> void* {
        off = (off + 255) & ~(size_t)255;
        void* p = (char*)d_ws + off;
        off += bytes;
        return p;
    };
    bf16_t* wtS1 = (bf16_t*)carve((size_t)27 * 256 * 128 * 2);
    bf16_t* wt11 = (bf16_t*)carve((size_t)27 * 128 * 128 * 2);
    bf16_t* wt12 = (bf16_t*)carve((size_t)27 * 128 * 128 * 2);
    bf16_t* wt21 = (bf16_t*)carve((size_t)27 * 128 * 128 * 2);
    bf16_t* wt22 = (bf16_t*)carve((size_t)27 * 128 * 128 * 2);
    bf16_t* wtS2 = (bf16_t*)carve((size_t)27 * 128 * 32 * 2);
    bf16_t* xbf  = (bf16_t*)carve((size_t)N0 * 256 * 2);
    float*  conv = (float*) carve((size_t)M1 * 128 * 4);   // also holds M2*32 (smaller)
    float*  f32r = (float*) carve((size_t)M1 * 128 * 4);
    bf16_t* bfA  = (bf16_t*)carve((size_t)M1 * 128 * 2);
    bf16_t* bfB  = (bf16_t*)carve((size_t)M1 * 128 * 2);
    float*  st   = (float*) carve((size_t)2 * 128 * 4);

    const dim3 blk(256);
    auto ceil256 = [](long long n) { return (unsigned)((n + 255) / 256); };

    // --- prep: weights (transpose + bf16) and x (bf16) ---
    wconv<<<ceil256((long long)27 * 256 * 128), blk, 0, stream>>>(w_s1, wtS1, 256, 128);
    wconv<<<ceil256((long long)27 * 128 * 128), blk, 0, stream>>>(w11,  wt11, 128, 128);
    wconv<<<ceil256((long long)27 * 128 * 128), blk, 0, stream>>>(w12,  wt12, 128, 128);
    wconv<<<ceil256((long long)27 * 128 * 128), blk, 0, stream>>>(w21,  wt21, 128, 128);
    wconv<<<ceil256((long long)27 * 128 * 128), blk, 0, stream>>>(w22,  wt22, 128, 128);
    wconv<<<ceil256((long long)27 * 128 * 32),  blk, 0, stream>>>(w_s2, wtS2, 128, 32);
    f2bf<<<ceil256((long long)N0 * 256), blk, 0, stream>>>(x, xbf, N0 * 256);

    // --- spconv1: 256 -> 128 (BM=64 keeps double-buffered LDS at 66KB) ---
    spconv_wmma<256, 128, 64><<<M1 / 64, blk, 0, stream>>>(xbf, nb1, wtS1, conv, M1);
    bn_stats<<<128, blk, 0, stream>>>(conv, M1, 128, st);
    bn_apply<<<ceil256((long long)M1 * 128), blk, 0, stream>>>(
        conv, st, g_s1, b_s1, nullptr, f32r, bfA, M1, 128);

    // --- block1 (BM=128: halves B-panel traffic, 8 wmma per B fragment) ---
    spconv_wmma<128, 128, 128><<<M1 / 128, blk, 0, stream>>>(bfA, nbs, wt11, conv, M1);
    bn_stats<<<128, blk, 0, stream>>>(conv, M1, 128, st);
    bn_apply<<<ceil256((long long)M1 * 128), blk, 0, stream>>>(
        conv, st, g11, b11, nullptr, nullptr, bfB, M1, 128);

    spconv_wmma<128, 128, 128><<<M1 / 128, blk, 0, stream>>>(bfB, nbs, wt12, conv, M1);
    bn_stats<<<128, blk, 0, stream>>>(conv, M1, 128, st);
    bn_apply<<<ceil256((long long)M1 * 128), blk, 0, stream>>>(
        conv, st, g12, b12, f32r, f32r, bfA, M1, 128);

    // --- block2 ---
    spconv_wmma<128, 128, 128><<<M1 / 128, blk, 0, stream>>>(bfA, nbs, wt21, conv, M1);
    bn_stats<<<128, blk, 0, stream>>>(conv, M1, 128, st);
    bn_apply<<<ceil256((long long)M1 * 128), blk, 0, stream>>>(
        conv, st, g21, b21, nullptr, nullptr, bfB, M1, 128);

    spconv_wmma<128, 128, 128><<<M1 / 128, blk, 0, stream>>>(bfB, nbs, wt22, conv, M1);
    bn_stats<<<128, blk, 0, stream>>>(conv, M1, 128, st);
    bn_apply<<<ceil256((long long)M1 * 128), blk, 0, stream>>>(
        conv, st, g22, b22, f32r, f32r, bfA, M1, 128);

    // --- spconv2: 128 -> 32 ---
    spconv_wmma<128, 32, 128><<<M2 / 128, blk, 0, stream>>>(bfA, nb2, wtS2, conv, M2);
    bn_stats<<<32, blk, 0, stream>>>(conv, M2, 32, st);
    bn_apply<<<ceil256((long long)M2 * 32), blk, 0, stream>>>(
        conv, st, g_s2, b_s2, nullptr, outp, nullptr, M2, 32);
}
